// SpectralSelfAttention_4071628997144
// MI455X (gfx1250) — compile-verified, matrix operand order checked by reading, not run on hardware
//
#include <hip/hip_runtime.h>
#include <hip/hip_bf16.h>
#include <stdint.h>

typedef __attribute__((ext_vector_type(16))) _Float16 v16h;
typedef __attribute__((ext_vector_type(8)))  float    v8f;

static constexpr int Bb = 16;
static constexpr int Nn = 1024;
static constexpr int Dd = 768;
static constexpr int Hh = 12;
static constexpr int HD = 64;

__device__ __forceinline__ v8f wmma16(v16h a, v16h b, v8f c) {
  // D = A(16x32 f16) * B(32x16 f16) + C(16x16 f32)
  return __builtin_amdgcn_wmma_f32_16x16x32_f16(false, a, false, b, (short)0, c,
                                                false, false);
}

// Load a 16x32 WMMA fragment from LDS. Works for A tiles stored [row][k]
// and for B tiles stored transposed [col][k] (contiguous K either way).
// Layout per ISA 7.12.2: lane m/n = lane%16; halves 0..7 = K kb..kb+7,
// halves 8..15 = K kb+16..kb+23, kb = (lane>=16 ? 8 : 0).
__device__ __forceinline__ v16h frag_ld(const _Float16* lds, int ld, int mj0, int k0) {
  int lane = threadIdx.x & 31;
  const _Float16* p = lds + (mj0 + (lane & 15)) * ld + k0 + ((lane >> 4) << 3);
  v16h f;
#pragma unroll
  for (int i = 0; i < 8; ++i) f[i] = p[i];
#pragma unroll
  for (int i = 0; i < 8; ++i) f[8 + i] = p[16 + i];
  return f;
}

// CDNA5 async DMA (saddr form): 16B global -> LDS, tracked by ASYNCcnt.
// SGPR 64-bit base + 32-bit unsigned VGPR byte offset (GVS addressing).
__device__ __forceinline__ void async_b128(uint32_t lds_off, uint32_t voff,
                                           const void* sbase) {
  asm volatile("global_load_async_to_lds_b128 %0, %1, %2"
               :: "v"(lds_off), "v"(voff), "s"((uint64_t)(uintptr_t)sbase)
               : "memory");
}
__device__ __forceinline__ void wait_async0() {
  asm volatile("s_wait_asynccnt 0x0" ::: "memory");
}
__device__ __forceinline__ uint32_t lds_off32(const void* p) {
  return (uint32_t)(uintptr_t)p;
}
// pack 4 f32 -> 4 f16 and store as one b64 to LDS (8B-aligned dst)
__device__ __forceinline__ void pack4_store(_Float16* dst, float4 v) {
  union { _Float16 h[4]; uint64_t u; } pk;
  pk.h[0] = (_Float16)v.x; pk.h[1] = (_Float16)v.y;
  pk.h[2] = (_Float16)v.z; pk.h[3] = (_Float16)v.w;
  *(uint64_t*)dst = pk.u;
}

// ---------------------------------------------------------------------------
// Kernel 1: qkv = x @ Wqkv + bqkv, then RoPE(q,k) via per-(b,n) wavelength,
// scatter to (B,H,N,HD) f16 workspaces. Block tile 128x64, wave tile 32x32.
// Grid: (B*N/128, 3*D/64), 256 threads.
// ---------------------------------------------------------------------------
__global__ __launch_bounds__(256) void qkv_rope_kernel(
    const float* __restrict__ x, const float* __restrict__ Wqkv,
    const float* __restrict__ bqkv, const float* __restrict__ wavelength,
    _Float16* __restrict__ qh, _Float16* __restrict__ kh,
    _Float16* __restrict__ vh) {
  __shared__ _Float16 ldsA[128 * 40];  // 128 rows x 32 K (pad 8)
  __shared__ _Float16 ldsBt[64 * 40];  // 64 cols x 32 K (B transposed)
  __shared__ float    outT[128 * 68];  // staged 128x64 result

  const int tid = threadIdx.x;
  const int lane = tid & 31;
  const int wv = tid >> 5;
  const int rr = (wv >> 1) * 32;     // wave row base in tile
  const int cc = (wv & 1) * 32;      // wave col base in tile
  const int gr0 = blockIdx.x * 128;  // flat (b*N+n) row base
  const int c0 = blockIdx.y * 64;    // qkv column base

  // fixed per-thread copy assignments (row constant -> immediate offsets)
  const int arow = tid >> 1, acg = (tid & 1) * 16;         // A: 2 thr/row, 16 cols
  const int bk = tid >> 3, bng = (tid & 7) * 8;            // B: 8 thr/row, 8 cols
  const float* xbase = x + gr0 * Dd;                       // block-uniform base

  v8f acc00 = {}, acc01 = {}, acc10 = {}, acc11 = {};

  for (int k0 = 0; k0 < Dd; k0 += 32) {
    __syncthreads();
    {  // A: 128x32 f32 -> f16, 4x float4 per thread
      const float4* src = (const float4*)(xbase + arow * Dd + k0 + acg);
      _Float16* dst = &ldsA[arow * 40 + acg];
#pragma unroll
      for (int i = 0; i < 4; ++i) pack4_store(dst + 4 * i, src[i]);
    }
    {  // B: 32x64 f32 -> f16, transposed, 2x float4 per thread
      const float4* src = (const float4*)(Wqkv + (k0 + bk) * (3 * Dd) + c0 + bng);
#pragma unroll
      for (int i = 0; i < 2; ++i) {
        float4 v = src[i];
        _Float16* d = &ldsBt[(bng + 4 * i) * 40 + bk];
        d[0] = (_Float16)v.x; d[40] = (_Float16)v.y;
        d[80] = (_Float16)v.z; d[120] = (_Float16)v.w;
      }
    }
    if (k0 + 32 < Dd) {  // prefetch next tiles (global_prefetch_b8)
      __builtin_prefetch(xbase + arow * Dd + k0 + 32 + acg, 0, 0);
      __builtin_prefetch(Wqkv + (k0 + 32 + bk) * (3 * Dd) + c0 + bng, 0, 0);
    }
    __syncthreads();
    v16h a0 = frag_ld(ldsA, 40, rr, 0);
    v16h a1 = frag_ld(ldsA, 40, rr + 16, 0);
    v16h b0 = frag_ld(ldsBt, 40, cc, 0);
    v16h b1 = frag_ld(ldsBt, 40, cc + 16, 0);
    acc00 = wmma16(a0, b0, acc00);
    acc01 = wmma16(a0, b1, acc01);
    acc10 = wmma16(a1, b0, acc10);
    acc11 = wmma16(a1, b1, acc11);
  }
  __syncthreads();
  {  // stage C fragments: m = r + 8*(lane>=16), n = lane&15
    int rb = rr + ((lane >> 4) << 3);
    int cb = cc + (lane & 15);
#pragma unroll
    for (int r = 0; r < 8; ++r) {
      outT[(rb + r) * 68 + cb] = acc00[r];
      outT[(rb + r) * 68 + cb + 16] = acc01[r];
      outT[(rb + 16 + r) * 68 + cb] = acc10[r];
      outT[(rb + 16 + r) * 68 + cb + 16] = acc11[r];
    }
  }
  __syncthreads();
  // bias + rotate pairs (p, p+32) + scatter. Column block = one head of q/k/v.
  const int part = blockIdx.y / Hh;  // 0=q 1=k 2=v
  const int h = blockIdx.y % Hh;
  _Float16* dst = (part == 0) ? qh : (part == 1) ? kh : vh;
#pragma unroll
  for (int i = 0; i < 16; ++i) {
    int idx = i * 256 + tid;
    int r = idx >> 5;  // row within tile (0..127)
    int p = idx & 31;  // pair index (half-dim)
    int gr = gr0 + r;
    int b = gr >> 10, n = gr & 1023;
    float t1 = outT[r * 68 + p] + bqkv[c0 + p];
    float t2 = outT[r * 68 + p + 32] + bqkv[c0 + p + 32];
    float o1 = t1, o2 = t2;
    if (part < 2) {
      float wl = wavelength[b * Nn + n];
      float invf = exp2f((float)p * (-13.2877124f / 32.0f));  // 10000^(-p/32)
      float ph = wl * invf;
      float s, c;
      __sincosf(ph, &s, &c);
      o1 = t1 * c - t2 * s;
      o2 = t2 * c + t1 * s;
    }
    int base = ((b * Hh + h) * Nn + n) * HD;
    dst[base + p] = (_Float16)o1;
    dst[base + p + 32] = (_Float16)o2;
  }
}

// ---------------------------------------------------------------------------
// Kernel 2: flash attention per (b,h). Grid: (N/64, B*H), 256 threads.
// Q/K tiles arrive via GLOBAL_LOAD_ASYNC_TO_LDS_B128 (ASYNCcnt, saddr mode).
// ---------------------------------------------------------------------------
__global__ __launch_bounds__(256) void attn_kernel(
    const _Float16* __restrict__ qh, const _Float16* __restrict__ kh,
    const _Float16* __restrict__ vh, const unsigned char* __restrict__ pm,
    _Float16* __restrict__ ao) {
  __shared__ _Float16 ldsQ[64 * 72];
  __shared__ _Float16 ldsK[64 * 72];   // [key][hd] == B^T for Q@K^T
  __shared__ _Float16 ldsVt[64 * 72];  // [hd][key] == B^T for P@V
  __shared__ _Float16 ldsP[64 * 72];
  __shared__ float    ldsS[64 * 68];
  __shared__ float    red[4 * 64], red2[4 * 64];
  __shared__ float    mrow[64], lrow[64], arw[64];

  const int tid = threadIdx.x;
  const int lane = tid & 31;
  const int wv = tid >> 5, wi = wv >> 1, wj = wv & 1;
  const int bh = blockIdx.y;
  const int b = bh / Hh, h = bh % Hh;
  const int qr0 = blockIdx.x * 64;
  const size_t base = (size_t)bh * Nn * HD;

  // per-thread DMA chunk: 512 x 16B per 64x64 tile
  const int cr = tid >> 3, cck = (tid & 7) * 16;  // row 0..31(x2), byte col
  {  // async copy Q tile (two rows-halves per thread)
    const char* qg = (const char*)(qh + base + qr0 * HD);
    async_b128(lds_off32(&ldsQ[cr * 72]) + cck, (uint32_t)(cr * 128 + cck), qg);
    async_b128(lds_off32(&ldsQ[(cr + 32) * 72]) + cck,
               (uint32_t)((cr + 32) * 128 + cck), qg);
  }
  if (tid < 64) { mrow[tid] = -3.0e38f; lrow[tid] = 0.f; }

  v8f o0 = {}; v8f o1 = {};
  const float scale = 0.125f;  // 1/sqrt(64)
  const int srow_i = tid & 63, seg = tid >> 6;

  for (int kc = 0; kc < Nn; kc += 64) {
    __syncthreads();
    {  // async copy K chunk [key][hd]
      const char* kg = (const char*)(kh + base + kc * HD);
      async_b128(lds_off32(&ldsK[cr * 72]) + cck, (uint32_t)(cr * 128 + cck), kg);
      async_b128(lds_off32(&ldsK[(cr + 32) * 72]) + cck,
                 (uint32_t)((cr + 32) * 128 + cck), kg);
      // V chunk transposed [hd][key] (needs VALU shuffle; dword loads)
      const uint32_t* gv = (const uint32_t*)(vh + base + kc * HD);
#pragma unroll
      for (int i = 0; i < 8; ++i) {
        int idx = i * 256 + tid;
        int key = idx >> 5, hp = idx & 31;
        uint32_t val = gv[key * 32 + hp];
        union { uint32_t u; _Float16 f[2]; } cv; cv.u = val;
        ldsVt[(2 * hp) * 72 + key] = cv.f[0];
        ldsVt[(2 * hp + 1) * 72 + key] = cv.f[1];
      }
    }
    wait_async0();
    __syncthreads();
    // S = Q @ K^T (K dim = HD = 64 -> two k-steps)
    v8f s0 = {}; v8f s1 = {};
    {
      v16h a0 = frag_ld(ldsQ, 72, 16 * wi, 0);
      v16h a1 = frag_ld(ldsQ, 72, 16 * wi, 32);
      v16h k0 = frag_ld(ldsK, 72, 32 * wj, 0);
      v16h k1 = frag_ld(ldsK, 72, 32 * wj, 32);
      v16h k2 = frag_ld(ldsK, 72, 32 * wj + 16, 0);
      v16h k3 = frag_ld(ldsK, 72, 32 * wj + 16, 32);
      s0 = wmma16(a0, k0, s0); s0 = wmma16(a1, k1, s0);
      s1 = wmma16(a0, k2, s1); s1 = wmma16(a1, k3, s1);
    }
    {  // stage scores with scale + pad mask
      int rbase = 16 * wi + ((lane >> 4) << 3);
      int cb = 32 * wj + (lane & 15);
      bool ok0 = pm[b * Nn + kc + cb] != 0;
      bool ok1 = pm[b * Nn + kc + cb + 16] != 0;
#pragma unroll
      for (int r = 0; r < 8; ++r) {
        ldsS[(rbase + r) * 68 + cb] = ok0 ? s0[r] * scale : -3.0e38f;
        ldsS[(rbase + r) * 68 + cb + 16] = ok1 ? s1[r] * scale : -3.0e38f;
      }
    }
    __syncthreads();
    {  // parallel online softmax: 4 threads per row, 16 cols each
      const float* srow = &ldsS[srow_i * 68 + seg * 16];
      float lm = -3.0e38f;
#pragma unroll
      for (int j = 0; j < 16; ++j) lm = fmaxf(lm, srow[j]);
      red[seg * 64 + srow_i] = lm;
    }
    __syncthreads();
    {
      float mnew = fmaxf(fmaxf(fmaxf(red[srow_i], red[64 + srow_i]),
                               fmaxf(red[128 + srow_i], red[192 + srow_i])),
                         mrow[srow_i]);
      const float* srow = &ldsS[srow_i * 68 + seg * 16];
      _Float16* prow = &ldsP[srow_i * 72 + seg * 16];
      float ps = 0.f;
#pragma unroll
      for (int j = 0; j < 16; ++j) {
        float e = __expf(srow[j] - mnew);
        ps += e;
        prow[j] = (_Float16)e;
      }
      red2[seg * 64 + srow_i] = ps;
    }
    __syncthreads();
    if (tid < 64) {  // finalize row stats
      float mold = mrow[tid];
      float mnew = fmaxf(fmaxf(fmaxf(red[tid], red[64 + tid]),
                               fmaxf(red[128 + tid], red[192 + tid])),
                         mold);
      float al = __expf(mold - mnew);
      lrow[tid] = lrow[tid] * al +
                  (red2[tid] + red2[64 + tid] + red2[128 + tid] + red2[192 + tid]);
      arw[tid] = al;
      mrow[tid] = mnew;
    }
    __syncthreads();
    {  // rescale running O, then O += P @ V
      int rbase = 16 * wi + ((lane >> 4) << 3);
#pragma unroll
      for (int r = 0; r < 8; ++r) {
        float al = arw[rbase + r];
        o0[r] *= al; o1[r] *= al;
      }
      v16h a0 = frag_ld(ldsP, 72, 16 * wi, 0);
      v16h a1 = frag_ld(ldsP, 72, 16 * wi, 32);
      v16h v0 = frag_ld(ldsVt, 72, 32 * wj, 0);
      v16h v1 = frag_ld(ldsVt, 72, 32 * wj, 32);
      v16h v2 = frag_ld(ldsVt, 72, 32 * wj + 16, 0);
      v16h v3 = frag_ld(ldsVt, 72, 32 * wj + 16, 32);
      o0 = wmma16(a0, v0, o0); o0 = wmma16(a1, v1, o0);
      o1 = wmma16(a0, v2, o1); o1 = wmma16(a1, v3, o1);
    }
  }
  __syncthreads();
  {  // normalize + store to (B, N, D) f16
    int rbase = 16 * wi + ((lane >> 4) << 3);
    int cb = 32 * wj + (lane & 15);
#pragma unroll
    for (int r = 0; r < 8; ++r) {
      int row = rbase + r;
      float inv = 1.0f / fmaxf(lrow[row], 1e-30f);
      int o = ((b * Nn + qr0 + row) * Dd) + h * HD;
      ao[o + cb] = (_Float16)(o0[r] * inv);
      ao[o + cb + 16] = (_Float16)(o1[r] * inv);
    }
  }
}

// ---------------------------------------------------------------------------
// Kernel 3: out = attn_out @ Wout + bout (f32 out). Block tile 128x64,
// wave tile 32x32. A tile arrives via async DMA. Grid: (B*N/128, D/64).
// ---------------------------------------------------------------------------
__global__ __launch_bounds__(256) void proj_kernel(
    const _Float16* __restrict__ ao, const float* __restrict__ Wout,
    const float* __restrict__ bout, float* __restrict__ out) {
  __shared__ _Float16 ldsA[128 * 40];
  __shared__ _Float16 ldsBt[64 * 40];
  const int tid = threadIdx.x;
  const int lane = tid & 31;
  const int wv = tid >> 5;
  const int rr = (wv >> 1) * 32;
  const int cc = (wv & 1) * 32;
  const int gr0 = blockIdx.x * 128;
  const int c0 = blockIdx.y * 64;
  const int ar = tid >> 1, ac = (tid & 1) * 16;  // A DMA: 2 chunks/thread
  const int bk = tid >> 3, bng = (tid & 7) * 8;  // B: 8 thr/row, 8 cols
  v8f acc00 = {}, acc01 = {}, acc10 = {}, acc11 = {};

  for (int k0 = 0; k0 < Dd; k0 += 32) {
    __syncthreads();
    {  // A: 128 rows x 64B via async b128 DMA (saddr = tile base)
      const char* ag = (const char*)(ao + gr0 * Dd + k0);
      async_b128(lds_off32(&ldsA[ar * 40]) + ac * 2,
                 (uint32_t)(ar * (Dd * 2) + ac * 2), ag);
      async_b128(lds_off32(&ldsA[ar * 40]) + ac * 2 + 16,
                 (uint32_t)(ar * (Dd * 2) + ac * 2 + 16), ag);
    }
    {  // B: 32x64 f32 -> f16, transposed
      const float4* src = (const float4*)(Wout + (k0 + bk) * Dd + c0 + bng);
#pragma unroll
      for (int i = 0; i < 2; ++i) {
        float4 v = src[i];
        _Float16* d = &ldsBt[(bng + 4 * i) * 40 + bk];
        d[0] = (_Float16)v.x; d[40] = (_Float16)v.y;
        d[80] = (_Float16)v.z; d[120] = (_Float16)v.w;
      }
    }
    if (k0 + 32 < Dd)
      __builtin_prefetch(Wout + (k0 + 32 + bk) * Dd + c0 + bng, 0, 0);
    wait_async0();
    __syncthreads();
    v16h a0 = frag_ld(ldsA, 40, rr, 0);
    v16h a1 = frag_ld(ldsA, 40, rr + 16, 0);
    v16h b0 = frag_ld(ldsBt, 40, cc, 0);
    v16h b1 = frag_ld(ldsBt, 40, cc + 16, 0);
    acc00 = wmma16(a0, b0, acc00);
    acc01 = wmma16(a0, b1, acc01);
    acc10 = wmma16(a1, b0, acc10);
    acc11 = wmma16(a1, b1, acc11);
  }
  int rb = rr + ((lane >> 4) << 3);
  int cb0 = c0 + cc + (lane & 15);
  float bias0 = bout[cb0];
  float bias1 = bout[cb0 + 16];
#pragma unroll
  for (int r = 0; r < 8; ++r) {
    int row0 = gr0 + rb + r;
    int row1 = row0 + 16;
    out[row0 * Dd + cb0] = acc00[r] + bias0;
    out[row0 * Dd + cb0 + 16] = acc01[r] + bias1;
    out[row1 * Dd + cb0] = acc10[r] + bias0;
    out[row1 * Dd + cb0 + 16] = acc11[r] + bias1;
  }
}

extern "C" void kernel_launch(void* const* d_in, const int* in_sizes, int n_in,
                              void* d_out, int out_size, void* d_ws,
                              size_t ws_size, hipStream_t stream) {
  const float* x = (const float*)d_in[0];
  const float* wavelength = (const float*)d_in[1];
  const unsigned char* pm = (const unsigned char*)d_in[2];
  const float* Wqkv = (const float*)d_in[3];
  const float* bqkv = (const float*)d_in[4];
  const float* Wout = (const float*)d_in[5];
  const float* bout = (const float*)d_in[6];
  float* out = (float*)d_out;

  const size_t QKH = (size_t)Bb * Hh * Nn * HD;  // per-tensor f16 elements
  _Float16* qh = (_Float16*)d_ws;
  _Float16* kh = qh + QKH;
  _Float16* vh = kh + QKH;
  _Float16* ao = vh + QKH;  // attn output (B,N,D) f16

  qkv_rope_kernel<<<dim3((Bb * Nn) / 128, (3 * Dd) / 64), 256, 0, stream>>>(
      x, Wqkv, bqkv, wavelength, qh, kh, vh);
  attn_kernel<<<dim3(Nn / 64, Bb * Hh), 256, 0, stream>>>(qh, kh, vh, pm, ao);
  proj_kernel<<<dim3((Bb * Nn) / 128, Dd / 64), 256, 0, stream>>>(ao, Wout,
                                                                  bout, out);
}
